// AdvancedChannelRankingLoss_29789893165635
// MI455X (gfx1250) — matrix operand order repaired and checked by reading, not compiled, not run on hardware
//
#include <hip/hip_runtime.h>
#include <hip/hip_bf16.h>
#include <math.h>

// ---------------- types for WMMA ----------------
typedef __bf16 bf16_t;
typedef bf16_t v16bf __attribute__((ext_vector_type(16)));
typedef bf16_t v4bf  __attribute__((ext_vector_type(4)));
typedef float  v8f   __attribute__((ext_vector_type(8)));
typedef float  f32x4 __attribute__((ext_vector_type(4)));
typedef unsigned int u32x4 __attribute__((ext_vector_type(4)));

union ABu { v16bf v; u32x4 u[2]; };

// ---------------- problem constants ----------------
#define CN   64          // channels
#define DN   1024        // embedding dim
#define KACT 4
#define ESTR 1032        // bf16 elems per LDS row (2064B row = 4-bank shift -> conflict-free b128)
#define TEMP_INV (1.0f/0.07f)

// LDS partition (dynamic shared)
#define OFF_EBF   0
#define SZ_EBF    (CN * ESTR * 2)            // 132096
#define OFF_GRAM  (OFF_EBF + SZ_EBF)         // 132096
#define SZ_GRAM   (CN * CN * 4)              // 16384
#define OFF_RN    (OFF_GRAM + SZ_GRAM)       // 148480
#define SZ_RN     (CN * 4)                   // 256
#define OFF_IDX   (OFF_RN + SZ_RN)           // 148736
#define SZ_IDX    256
#define OFF_RED   (OFF_IDX + SZ_IDX)         // 148992
#define SZ_RED    256
#define SMEM_BYTES (OFF_RED + SZ_RED)        // 149248 -> 2 workgroups / 320KB WGP

__device__ __constant__ int c_offdiag[12] = {1,2,3,4,6,7,8,9,11,12,13,14};

// =====================================================================
// Kernel 0: zero the scalar accumulators in workspace
// =====================================================================
__global__ void init_acc_kernel(float* acc) {
    if (threadIdx.x < 8) acc[threadIdx.x] = 0.0f;
}

// =====================================================================
// Kernel 1: score / margin / topk / spatial / network terms.
// One 64-thread block (2 waves) per batch row.
// =====================================================================
__global__ void small_losses_kernel(const float* __restrict__ pred,
                                    const int*   __restrict__ labels,
                                    const float* __restrict__ cpos,
                                    const float* __restrict__ conn,
                                    float* __restrict__ acc,
                                    float bInv) {
    __shared__ float sp[CN], sm[CN], px[CN], py[CN], pz[CN];
    __shared__ int   slab[CN];
    __shared__ float rb[CN], rs[CN], rn[CN];

    const int b = blockIdx.x;
    const int t = threadIdx.x;

    float p   = pred[b * CN + t];
    int   lab = labels[b * CN + t];
    sp[t] = p;
    slab[t] = lab;
    sm[t] = (p > 0.5f) ? 1.0f : 0.0f;
    px[t] = cpos[t * 3 + 0];
    py[t] = cpos[t * 3 + 1];
    pz[t] = cpos[t * 3 + 2];
    rb[t] = lab ? -logf(p) : -log1pf(-p);
    __syncthreads();

    // spatial: m_t * sum_j m_j * dist(t,j)
    float s_sp = 0.0f;
    if (sm[t] > 0.0f) {
        float x = px[t], y = py[t], z = pz[t];
        for (int j = 0; j < CN; ++j) {
            if (j != t && sm[j] > 0.0f) {
                float dx = x - px[j], dy = y - py[j], dz = z - pz[j];
                s_sp += sqrtf(dx * dx + dy * dy + dz * dz);
            }
        }
    }
    rs[t] = s_sp;

    // network: p_t * sum_j W[t][j] * p_j
    float s_net = 0.0f;
    for (int j = 0; j < CN; ++j) s_net += conn[t * CN + j] * sp[j];
    rn[t] = s_net * p;
    __syncthreads();

    if (t == 0) {
        float bce = 0.f, asum = 0.f, psum = 0.f, spat = 0.f, net = 0.f;
        int nact = 0, nm = 0;
        for (int j = 0; j < CN; ++j) {
            bce += rb[j]; psum += sp[j];
            if (slab[j]) { asum += sp[j]; nact++; }
            spat += rs[j]; net += rn[j];
            if (sm[j] > 0.0f) nm++;
        }
        float amean = asum / (float)nact;
        float imean = (psum - asum) / (float)(CN - nact);
        float marg = fmaxf(0.0f, 0.15f - (amean - imean));
        // topk (K=4, ties -> lowest index like lax.top_k)
        unsigned long long used = 0ull;
        int inter = 0;
        for (int k = 0; k < KACT; ++k) {
            int best = -1; float bv = -1e30f;
            for (int j = 0; j < CN; ++j)
                if (!((used >> j) & 1ull) && sp[j] > bv) { bv = sp[j]; best = j; }
            used |= (1ull << best);
            if (slab[best]) inter++;
        }
        float uni = (float)(KACT + nact - inter);
        float tk = 1.0f - (float)inter / (uni + 1e-8f);
        float fn = (float)nm;
        float avg = (nm >= 2) ? (spat / fmaxf(fn * (fn - 1.0f), 1.0f)) : 0.0f;
        float coh = net / ((float)CN * (float)CN);

        atomicAdd(acc + 0, bce * bInv * (1.0f / (float)CN));
        atomicAdd(acc + 1, marg * bInv);
        atomicAdd(acc + 2, tk * bInv);
        atomicAdd(acc + 4, avg * bInv);
        atomicAdd(acc + 5, -coh * bInv);
    }
}

// =====================================================================
// Kernel 2: contrastive term.
// One 256-thread (8 wave32) block per batch element.
//   Phase 1: stream E[b] (64x1024 f32, non-temporal) -> LDS bf16
//   Phase 2: full 64x64 Gram via v_wmma_f32_16x16x32_bf16 (16 tiles, 2/wave)
//   Phase 3: rnorm from Gram diagonal; 12-way InfoNCE rows; reduce.
// =====================================================================
__global__ void contrastive_kernel(const float* __restrict__ emb,
                                   const int*   __restrict__ labels,
                                   float* __restrict__ acc,
                                   float bInv) {
    extern __shared__ char smem[];
    bf16_t* ebf   = (bf16_t*)(smem + OFF_EBF);
    float*  gram  = (float*)(smem + OFF_GRAM);
    float*  rnorm = (float*)(smem + OFF_RN);
    int*    aidx  = (int*)(smem + OFF_IDX);        // [4]
    int*    iidx  = aidx + KACT;                   // [60]
    float*  red   = (float*)(smem + OFF_RED);      // [64]

    const int b    = blockIdx.x;
    const int tid  = threadIdx.x;
    const int lane = tid & 31;
    // wave index is uniform per wave -> force into SGPR so the tile loop is SALU
    const int wave = __builtin_amdgcn_readfirstlane(tid >> 5);

    // ---- Phase 1: global f32 (non-temporal stream) -> LDS bf16 ----------
    // 8 b128 NT loads in flight per thread, then convert + ds_store.
    const f32x4* src = (const f32x4*)(emb + (size_t)b * (CN * DN));
    for (int i = 0; i < (CN * DN / 4) / 256; i += 8) {   // 64 iters, 8 at a time
        f32x4 f[8];
        #pragma unroll
        for (int u = 0; u < 8; ++u)
            f[u] = __builtin_nontemporal_load(src + (i + u) * 256 + tid);
        #pragma unroll
        for (int u = 0; u < 8; ++u) {
            int idx4 = (i + u) * 256 + tid;
            int row  = idx4 >> 8;          // 256 float4 per row
            int kc   = (idx4 & 255) << 2;
            v4bf h = { (bf16_t)f[u][0], (bf16_t)f[u][1], (bf16_t)f[u][2], (bf16_t)f[u][3] };
            *(v4bf*)(ebf + row * ESTR + kc) = h;
        }
    }
    __syncthreads();

    // ---- Phase 2: 64x64 Gram via WMMA bf16 ------------------------------
    // A layout (16x32 bf16): lane<16 -> row=lane,    K {0..15}
    //                        lane>=16-> row=lane-16, K {16..31}
    // B layout (32x16 bf16): lane<16 -> col=lane,    K {0..15}
    //                        lane>=16-> col=lane-16, K {16..31}
    const int lhalf = (lane < 16) ? 0 : 1;
    #pragma unroll
    for (int tt = 0; tt < 2; ++tt) {
        const int t  = wave + tt * 8;              // scalar tile id
        const int mt = t >> 2, nt = t & 3;
        const int mrow = mt * 16 + (lane & 15);
        const int ncol = nt * 16 + (lane & 15);
        const int abase = mrow * ESTR + (lhalf ? 16 : 0);
        const int bbase = ncol * ESTR + (lhalf ? 16 : 0);
        v8f cacc = {};
        #pragma unroll 4
        for (int kb = 0; kb < DN; kb += 32) {
            ABu a, bb;
            a.u[0]  = *(const u32x4*)(ebf + abase + kb);        // K {0..7}(+16)
            a.u[1]  = *(const u32x4*)(ebf + abase + kb + 8);    // K {8..15}(+16)
            bb.u[0] = *(const u32x4*)(ebf + bbase + kb);        // K {0..7}(+16)
            bb.u[1] = *(const u32x4*)(ebf + bbase + kb + 8);    // K {8..15}(+16)
            cacc = __builtin_amdgcn_wmma_f32_16x16x32_bf16(
                false, a.v, false, bb.v, (short)0, cacc, false, false);
        }
        const int m0 = mt * 16 + (lhalf ? 8 : 0);
        const int n  = nt * 16 + (lane & 15);
        #pragma unroll
        for (int r = 0; r < 8; ++r) gram[(m0 + r) * CN + n] = cacc[r];
    }
    __syncthreads();

    // ---- Phase 3: norms, index lists, InfoNCE ---------------------------
    if (tid < CN) rnorm[tid] = rsqrtf(fmaxf(gram[tid * CN + tid], 1e-24f));
    if (tid == 0) {
        int na = 0, ni = 0;
        for (int c = 0; c < CN; ++c) {
            int l = labels[b * CN + c];
            if (l) { if (na < KACT) aidx[na] = c; na++; }
            else   { iidx[ni++] = c; }
        }
    }
    if (tid < CN) red[tid] = 0.0f;
    __syncthreads();

    if (tid < 12) {
        const int p  = tid;
        const int od = c_offdiag[p];
        const int ak = aidx[od >> 2], aj = aidx[od & 3];
        float l0 = gram[ak * CN + aj] * rnorm[ak] * rnorm[aj] * TEMP_INV;
        float mx = l0;
        float lg[21];
        lg[0] = l0;
        #pragma unroll
        for (int n = 1; n <= 20; ++n) {
            int f = p * 20 + (n - 1);
            int a = aidx[f / 60];
            int c = iidx[f % 60];
            float v = gram[a * CN + c] * rnorm[a] * rnorm[c] * TEMP_INV;
            lg[n] = v;
            mx = fmaxf(mx, v);
        }
        float s = 0.0f;
        #pragma unroll
        for (int n = 0; n <= 20; ++n) s += expf(lg[n] - mx);
        red[p] = logf(s) + mx - l0;      // ce for this positive pair
    }
    __syncthreads();
    if (tid == 0) {
        float ce = 0.0f;
        for (int p = 0; p < 12; ++p) ce += red[p];
        atomicAdd(acc + 3, ce * (bInv / 12.0f));
    }
}

// =====================================================================
// Kernel 3: weighted combine -> scalar output
// =====================================================================
__global__ void combine_kernel(const float* __restrict__ acc, float* __restrict__ out) {
    if (threadIdx.x == 0) {
        out[0] = 3.0f * acc[0]   // score
               + 1.0f * acc[1]   // margin
               + 2.0f * acc[2]   // topk
               + 1.0f * acc[3]   // contrastive
               + 0.5f * acc[4]   // spatial
               + 0.5f * acc[5];  // network
    }
}

// =====================================================================
extern "C" void kernel_launch(void* const* d_in, const int* in_sizes, int n_in,
                              void* d_out, int out_size, void* d_ws, size_t ws_size,
                              hipStream_t stream) {
    const float* pred   = (const float*)d_in[0];
    const int*   labels = (const int*)d_in[1];
    const float* emb    = (const float*)d_in[2];
    const float* cpos   = (const float*)d_in[3];
    const float* conn   = (const float*)d_in[4];
    float* out = (float*)d_out;
    float* acc = (float*)d_ws;

    const int B = in_sizes[0] / CN;          // 2048
    const float bInv = 1.0f / (float)B;

    init_acc_kernel<<<1, 32, 0, stream>>>(acc);
    small_losses_kernel<<<B, CN, 0, stream>>>(pred, labels, cpos, conn, acc, bInv);
    contrastive_kernel<<<B, 256, SMEM_BYTES, stream>>>(emb, labels, acc, bInv);
    combine_kernel<<<1, 1, 0, stream>>>(acc, out);
}
